// SetAbstractionModule_80444737454675
// MI455X (gfx1250) — compile-verified, matrix operand order checked by reading, not run on hardware
//
#include <hip/hip_runtime.h>

#define NPTS   4096
#define NBAT   16
#define NSAMP  1024
#define KNEI   64
#define CIN    64
#define RAD2   0.04f

typedef __attribute__((ext_vector_type(16))) _Float16 v16h;
typedef __attribute__((ext_vector_type(8)))  float    v8f;

// ---------------------------------------------------------------------------
// 1) Farthest point sampling: 1 block per cloud, dmin held in registers.
// ---------------------------------------------------------------------------
__global__ __launch_bounds__(1024) void fps_kernel(const float* __restrict__ pos,
                                                   int* __restrict__ fpsIdx) {
  const int b   = blockIdx.x;
  const float* p = pos + (size_t)b * NPTS * 3;
  const int tid = threadIdx.x;
  const int lane = tid & 31, wv = tid >> 5;

  __shared__ float redV[32];
  __shared__ int   redI[32];
  __shared__ float cbx, cby, cbz;

  float px[4], py[4], pz[4], dmin[4];
#pragma unroll
  for (int q = 0; q < 4; ++q) {
    int i = tid + q * 1024;
    px[q] = p[i * 3 + 0]; py[q] = p[i * 3 + 1]; pz[q] = p[i * 3 + 2];
  }
  const float c0x = p[0], c0y = p[1], c0z = p[2];
#pragma unroll
  for (int q = 0; q < 4; ++q) {
    float dx = px[q] - c0x, dy = py[q] - c0y, dz = pz[q] - c0z;
    dmin[q] = dx * dx + dy * dy + dz * dz;
  }
  if (tid == 0) fpsIdx[b * NSAMP] = 0;

  for (int step = 1; step < NSAMP; ++step) {
    // argmax(dmin), first index wins ties (matches jnp.argmax)
    float bv = -1.f; int bi = 0x7fffffff;
#pragma unroll
    for (int q = 0; q < 4; ++q) {
      int i = tid + q * 1024;
      float v = dmin[q];
      if (v > bv || (v == bv && i < bi)) { bv = v; bi = i; }
    }
#pragma unroll
    for (int off = 16; off > 0; off >>= 1) {
      float ov = __shfl_down(bv, off);
      int   oi = __shfl_down(bi, off);
      if (ov > bv || (ov == bv && oi < bi)) { bv = ov; bi = oi; }
    }
    if (lane == 0) { redV[wv] = bv; redI[wv] = bi; }
    __syncthreads();
    if (wv == 0) {
      bv = redV[lane]; bi = redI[lane];
#pragma unroll
      for (int off = 16; off > 0; off >>= 1) {
        float ov = __shfl_down(bv, off);
        int   oi = __shfl_down(bi, off);
        if (ov > bv || (ov == bv && oi < bi)) { bv = ov; bi = oi; }
      }
      if (lane == 0) {
        fpsIdx[b * NSAMP + step] = bi;
        cbx = p[bi * 3 + 0]; cby = p[bi * 3 + 1]; cbz = p[bi * 3 + 2];
      }
    }
    __syncthreads();
    const float cx = cbx, cy = cby, cz = cbz;
#pragma unroll
    for (int q = 0; q < 4; ++q) {
      float dx = px[q] - cx, dy = py[q] - cy, dz = pz[q] - cz;
      float d2 = dx * dx + dy * dy + dz * dz;
      dmin[q] = fminf(dmin[q], d2);
    }
  }
}

// ---------------------------------------------------------------------------
// 2) Ball query: K nearest within radius. 1 thread per center, replace-worst
//    list of 64 in LDS laid out [slot][tid] (conflict-free across 64 banks).
// ---------------------------------------------------------------------------
__global__ __launch_bounds__(64) void ballq_kernel(const float* __restrict__ pos,
                                                   const int* __restrict__ fpsIdx,
                                                   int* __restrict__ nbr,
                                                   int* __restrict__ cntOut) {
  const int tid = threadIdx.x;
  const int gid = blockIdx.x * 64 + tid;      // 0..16383
  const int b   = gid >> 10;
  const int ci  = fpsIdx[gid];
  const float* p = pos + (size_t)b * NPTS * 3;
  const float cx = p[ci * 3 + 0], cy = p[ci * 3 + 1], cz = p[ci * 3 + 2];

  __shared__ float sd[KNEI * 64];
  __shared__ int   si[KNEI * 64];

  int cnt = 0, wsl = 0;
  float wvv = -1.f;
  for (int j = 0; j < NPTS; ++j) {
    float dx = p[j * 3 + 0] - cx, dy = p[j * 3 + 1] - cy, dz = p[j * 3 + 2] - cz;
    float d2 = dx * dx + dy * dy + dz * dz;
    if (d2 <= RAD2) {
      if (cnt < KNEI) {
        sd[cnt * 64 + tid] = d2; si[cnt * 64 + tid] = j; cnt++;
        if (cnt == KNEI) {
          wvv = -1.f;
          for (int s2 = 0; s2 < KNEI; ++s2) {
            float v = sd[s2 * 64 + tid];
            if (v > wvv) { wvv = v; wsl = s2; }
          }
        }
      } else if (d2 < wvv) {           // strict < keeps earlier index on ties
        sd[wsl * 64 + tid] = d2; si[wsl * 64 + tid] = j;
        wvv = -1.f;
        for (int s2 = 0; s2 < KNEI; ++s2) {
          float v = sd[s2 * 64 + tid];
          if (v > wvv) { wvv = v; wsl = s2; }
        }
      }
    }
  }
  for (int k = 0; k < KNEI; ++k)
    nbr[(size_t)gid * KNEI + k] = (k < cnt) ? si[k * 64 + tid] : ci;
  cntOut[gid] = cnt;
}

// ---------------------------------------------------------------------------
// 3) Pre-swizzle f32 weights into per-lane WMMA B-fragment layout (f16).
//    Tile (kt,nt) = 32x16; lane L holds column N = nt*16+(L&15), 16 halves
//    with K permutation kk = (h<8 ? grp*8+h : 16+grp*8+(h-8)), grp = L>=16.
// ---------------------------------------------------------------------------
__global__ void prep_weights(const float* __restrict__ W1, const float* __restrict__ W2,
                             const float* __restrict__ W3, _Float16* __restrict__ w1s,
                             _Float16* __restrict__ w2s, _Float16* __restrict__ w3s) {
  const int n1 = 3 * 4 * 512, n2 = 2 * 4 * 512, n3 = 2 * 8 * 512;
  int idx = blockIdx.x * blockDim.x + threadIdx.x;
  if (idx >= n1 + n2 + n3) return;
  const float* W; _Float16* out; int NT, d0, d1, r;
  if (idx < n1)            { W = W1; out = w1s; NT = 4; d0 = 67; d1 = 64;  r = idx; }
  else if (idx < n1 + n2)  { W = W2; out = w2s; NT = 4; d0 = 64; d1 = 64;  r = idx - n1; }
  else                     { W = W3; out = w3s; NT = 8; d0 = 64; d1 = 128; r = idx - n1 - n2; }
  int tile = r >> 9, rem = r & 511;
  int lane = rem >> 4, h = rem & 15;
  int kt = tile / NT, nt = tile % NT;
  int grp = lane >> 4;
  int kk = (h < 8) ? (grp * 8 + h) : (16 + grp * 8 + (h - 8));
  int K = kt * 32 + kk;
  int n = nt * 16 + (lane & 15);
  float v = (K < d0) ? W[K * d1 + n] : 0.f;
  out[r] = (_Float16)v;
}

// ---------------------------------------------------------------------------
// 4) WMMA MLP + masked row-max. 1 block = 1 center; 4 waves = 4 M-tiles.
// ---------------------------------------------------------------------------
__global__ __launch_bounds__(128) void mlp_kernel(
    const float* __restrict__ x, const float* __restrict__ pos,
    const int* __restrict__ fpsIdx, const int* __restrict__ nbr,
    const int* __restrict__ cntArr,
    const _Float16* __restrict__ w1s, const _Float16* __restrict__ w2s,
    const _Float16* __restrict__ w3s,
    const float* __restrict__ b1, const float* __restrict__ b2,
    const float* __restrict__ b3, float* __restrict__ out) {
  const int s = blockIdx.x;        // global center id, 0..16383
  const int b = s >> 10;
  const int tid = threadIdx.x;
  const int w = tid >> 5, lane = tid & 31;
  const int m = lane & 15, grp = lane >> 4;

  __shared__ _Float16 F[64 * 96];     // input features, K padded 67->96
  __shared__ _Float16 G[64 * 64];     // layer-1 output
  __shared__ _Float16 H[64 * 64];     // layer-2 output
  __shared__ float    O3[64 * 128];   // layer-3 output (f32)
  __shared__ int nbrS[64];
  __shared__ float cen[3];
  __shared__ int cntS;

  if (tid == 0) cntS = cntArr[s];
  if (tid < 64) nbrS[tid] = nbr[(size_t)s * KNEI + tid];
  const int ci = fpsIdx[s];
  if (tid < 3) cen[tid] = pos[((size_t)b * NPTS + ci) * 3 + tid];
  __syncthreads();

  { // build F: 2 threads per neighbor row
    int k = tid >> 1, half = tid & 1;
    int j = nbrS[k];
    const float* xr = x + ((size_t)b * NPTS + j) * CIN;
    if (half == 0) {
      for (int c = 0; c < 48; ++c) F[k * 96 + c] = (_Float16)xr[c];
    } else {
      for (int c = 48; c < 64; ++c) F[k * 96 + c] = (_Float16)xr[c];
      const float* pr = pos + ((size_t)b * NPTS + j) * 3;
      for (int d = 0; d < 3; ++d) F[k * 96 + 64 + d] = (_Float16)(pr[d] - cen[d]);
      for (int c = 67; c < 96; ++c) F[k * 96 + c] = (_Float16)0.f;
    }
  }
  __syncthreads();

  // ---- Layer 1: [64x96] x [96x64] + b1, relu -> G ----
  for (int nt = 0; nt < 4; ++nt) {
    v8f acc = {0.f, 0.f, 0.f, 0.f, 0.f, 0.f, 0.f, 0.f};
    for (int kt = 0; kt < 3; ++kt) {
      v16h a;
#pragma unroll
      for (int h = 0; h < 16; ++h) {
        int kk = (h < 8) ? (grp * 8 + h) : (16 + grp * 8 + (h - 8));
        a[h] = F[(w * 16 + m) * 96 + kt * 32 + kk];
      }
      v16h bf = *(const v16h*)(w1s + (kt * 4 + nt) * 512 + lane * 16);
      acc = __builtin_amdgcn_wmma_f32_16x16x32_f16(false, a, false, bf, (short)0, acc,
                                                   false, false);
    }
    float bias = b1[nt * 16 + m];
    int rbase = w * 16 + grp * 8;
#pragma unroll
    for (int j2 = 0; j2 < 8; ++j2)
      G[(rbase + j2) * 64 + nt * 16 + m] = (_Float16)fmaxf(acc[j2] + bias, 0.f);
  }
  __syncthreads();

  // ---- Layer 2: [64x64] x [64x64] + b2, relu -> H ----
  for (int nt = 0; nt < 4; ++nt) {
    v8f acc = {0.f, 0.f, 0.f, 0.f, 0.f, 0.f, 0.f, 0.f};
    for (int kt = 0; kt < 2; ++kt) {
      v16h a;
#pragma unroll
      for (int h = 0; h < 16; ++h) {
        int kk = (h < 8) ? (grp * 8 + h) : (16 + grp * 8 + (h - 8));
        a[h] = G[(w * 16 + m) * 64 + kt * 32 + kk];
      }
      v16h bf = *(const v16h*)(w2s + (kt * 4 + nt) * 512 + lane * 16);
      acc = __builtin_amdgcn_wmma_f32_16x16x32_f16(false, a, false, bf, (short)0, acc,
                                                   false, false);
    }
    float bias = b2[nt * 16 + m];
    int rbase = w * 16 + grp * 8;
#pragma unroll
    for (int j2 = 0; j2 < 8; ++j2)
      H[(rbase + j2) * 64 + nt * 16 + m] = (_Float16)fmaxf(acc[j2] + bias, 0.f);
  }
  __syncthreads();

  // ---- Layer 3: [64x64] x [64x128] -> O3 (bias folded after max) ----
  for (int nt = 0; nt < 8; ++nt) {
    v8f acc = {0.f, 0.f, 0.f, 0.f, 0.f, 0.f, 0.f, 0.f};
    for (int kt = 0; kt < 2; ++kt) {
      v16h a;
#pragma unroll
      for (int h = 0; h < 16; ++h) {
        int kk = (h < 8) ? (grp * 8 + h) : (16 + grp * 8 + (h - 8));
        a[h] = H[(w * 16 + m) * 64 + kt * 32 + kk];
      }
      v16h bf = *(const v16h*)(w3s + (kt * 8 + nt) * 512 + lane * 16);
      acc = __builtin_amdgcn_wmma_f32_16x16x32_f16(false, a, false, bf, (short)0, acc,
                                                   false, false);
    }
    int rbase = w * 16 + grp * 8;
#pragma unroll
    for (int j2 = 0; j2 < 8; ++j2)
      O3[(rbase + j2) * 128 + nt * 16 + m] = acc[j2];
  }
  __syncthreads();

  // ---- masked max over valid neighbor rows, + b3 ----
  {
    const int c = tid;               // 0..127 output channels
    const int cc = cntS;             // >= 1 always (center in its own ball)
    float mx = -__builtin_inff();
    for (int r = 0; r < cc; ++r) mx = fmaxf(mx, O3[r * 128 + c]);
    out[(size_t)s * 128 + c] = mx + b3[c];
  }
  if (tid < 3)
    out[(size_t)NBAT * NSAMP * 128 + (size_t)s * 3 + tid] = cen[tid];
  if (tid == 3)
    out[(size_t)NBAT * NSAMP * 128 + (size_t)NBAT * NSAMP * 3 + s] = (float)b;
}

// ---------------------------------------------------------------------------
extern "C" void kernel_launch(void* const* d_in, const int* in_sizes, int n_in,
                              void* d_out, int out_size, void* d_ws, size_t ws_size,
                              hipStream_t stream) {
  const float* x   = (const float*)d_in[0];
  const float* pos = (const float*)d_in[1];
  // d_in[2] = batch (int64) -- values derivable, unused
  const float* W1 = (const float*)d_in[3];
  const float* b1 = (const float*)d_in[4];
  const float* W2 = (const float*)d_in[5];
  const float* b2 = (const float*)d_in[6];
  const float* W3 = (const float*)d_in[7];
  const float* b3 = (const float*)d_in[8];
  float* out = (float*)d_out;

  char* ws = (char*)d_ws;
  int* fpsIdx = (int*)(ws);                          //   64 KB
  int* nbr    = (int*)(ws + 65536);                  //    4 MB
  int* cnt    = (int*)(ws + 65536 + 4194304);        //   64 KB
  _Float16* w1s = (_Float16*)(ws + 4325376);         //   12 KB (swizzled)
  _Float16* w2s = (_Float16*)(ws + 4337664);         //    8 KB
  _Float16* w3s = (_Float16*)(ws + 4345856);         //   16 KB

  prep_weights<<<(18432 + 255) / 256, 256, 0, stream>>>(W1, W2, W3, w1s, w2s, w3s);
  fps_kernel<<<NBAT, 1024, 0, stream>>>(pos, fpsIdx);
  ballq_kernel<<<(NBAT * NSAMP) / 64, 64, 0, stream>>>(pos, fpsIdx, nbr, cnt);
  mlp_kernel<<<NBAT * NSAMP, 128, 0, stream>>>(x, pos, fpsIdx, nbr, cnt,
                                               w1s, w2s, w3s, b1, b2, b3, out);
}
